// Block_10453950398694
// MI455X (gfx1250) — matrix operand order repaired and checked by reading, not compiled, not run on hardware
//
#include <hip/hip_runtime.h>

// ---------------------------------------------------------------------------
// Transformer block (pre-LN MHA + pre-LN FFN) for MI455X (gfx1250, wave32).
// All GEMMs run on v_wmma_f32_16x16x32_bf16 with 64x32 wave tiles
// (8 WMMA accumulators, no spills). Epilogue variants are compile-time
// template specializations (no runtime flag branches). Attention: VALU + LDS.
// ---------------------------------------------------------------------------

typedef __attribute__((ext_vector_type(16))) __bf16 v16bf;
typedef __attribute__((ext_vector_type(8)))  float  v8f;

constexpr int kB   = 32;
constexpr int kT   = 512;
constexpr int kD   = 768;
constexpr int kH   = 12;
constexpr int kE   = 6;
constexpr int kDFF = 4 * kD;          // 3072
constexpr int kBT  = kB * kT;         // 16384
constexpr int kNQ  = 256;             // padded 3*H*E = 216 -> 256
constexpr int kKO  = 96;              // padded H*E = 72 -> 96 (mult of 32)
constexpr float kEPS = 1e-5f;

union FragU { v16bf v; uint4 u[2]; };

// A fragment (16x32 bf16, row-major source, row stride = K elements).
// Per ISA: lane half h, element j: k = (j>>3)*16 + h*8 + (j&7)
// -> two contiguous 16B chunks at k0 + h*8 and k0 + 16 + h*8.
__device__ inline v16bf load_a_frag(const __bf16* A, int K, int row0, int k0, int lane) {
    const __bf16* p = A + (size_t)(row0 + (lane & 15)) * K + k0 + ((lane >> 4) << 3);
    FragU t;
    t.u[0] = *(const uint4*)(p);
    t.u[1] = *(const uint4*)(p + 16);
    return t.v;
}

// B fragment (32x16 bf16) loaded from B^T stored [N x K] row-major.
// Per ISA: lane half h, element j: k = (j>>3)*16 + h*8 + (j&7)
// -> two contiguous 16B chunks at k0 + h*8 and k0 + 16 + h*8.
__device__ inline v16bf load_b_frag(const __bf16* Bt, int K, int col0, int k0, int lane) {
    const __bf16* p = Bt + (size_t)(col0 + (lane & 15)) * K + k0 + ((lane >> 4) << 3);
    FragU t;
    t.u[0] = *(const uint4*)(p);
    t.u[1] = *(const uint4*)(p + 16);
    return t.v;
}

// ---------------------------------------------------------------------------
// Generic bf16 WMMA GEMM: C[M,N] = A[M,K] * B[K,N] (+bias) (relu) (+residual)
// A row-major bf16 (lda=K), Bt = B^T row-major bf16 [N x K] (ldb=K).
// Block: 128 threads = 4 waves in 2x2; each wave owns a 64x32 tile
// (4x2 WMMA accumulators). Grid: (M/128, N/64). K multiple of 32.
// Epilogue behavior fixed at compile time (no pointer null-check branches).
// ---------------------------------------------------------------------------
template <bool HAS_BIAS, bool HAS_RES, bool RELU, bool OUT_BF16>
__global__ void __launch_bounds__(128, 1)
gemm_bf16_wmma(const __bf16* __restrict__ A,
               const __bf16* __restrict__ Bt,
               const float*  __restrict__ bias,
               const float*  __restrict__ resid,
               float* __restrict__ outF,
               __bf16* __restrict__ outB,
               int M, int N, int K) {
    const int lane = threadIdx.x & 31;
    const int wave = threadIdx.x >> 5;                 // 0..3
    const int m0 = blockIdx.x * 128 + (wave >> 1) * 64;
    const int n0 = blockIdx.y * 64 + (wave & 1) * 32;

    v8f acc[4][2] = {};

    for (int k0 = 0; k0 < K; k0 += 32) {
        v16bf a[4], b[2];
        #pragma unroll
        for (int i = 0; i < 4; ++i) a[i] = load_a_frag(A, K, m0 + 16 * i, k0, lane);
        #pragma unroll
        for (int i = 0; i < 2; ++i) b[i] = load_b_frag(Bt, K, n0 + 16 * i, k0, lane);
        #pragma unroll
        for (int mi = 0; mi < 4; ++mi)
            #pragma unroll
            for (int ni = 0; ni < 2; ++ni)
                acc[mi][ni] = __builtin_amdgcn_wmma_f32_16x16x32_bf16(
                    false, a[mi], false, b[ni], (short)0, acc[mi][ni], false, false);
    }

    // C/D layout: VGPR r -> M = r + (lane>>4)*8, N = lane&15 (within 16x16 tile)
    const int nl   = lane & 15;
    const int half = lane >> 4;
    #pragma unroll
    for (int mi = 0; mi < 4; ++mi) {
        #pragma unroll
        for (int ni = 0; ni < 2; ++ni) {
            const int nn = n0 + ni * 16 + nl;
            float bv = 0.0f;
            if (HAS_BIAS) bv = bias[nn];
            #pragma unroll
            for (int r = 0; r < 8; ++r) {
                const int mm = m0 + mi * 16 + half * 8 + r;
                float v = acc[mi][ni][r];
                if (HAS_BIAS) v += bv;
                if (RELU) v = fmaxf(v, 0.0f);
                const size_t idx = (size_t)mm * N + nn;
                if (HAS_RES) v += resid[idx];
                if (OUT_BF16) outB[idx] = (__bf16)v;
                else          outF[idx] = v;
            }
        }
    }
}

// ---------------------------------------------------------------------------
// LayerNorm (biased var) fp32 in -> bf16 out. One block (256 thr) per row.
// ---------------------------------------------------------------------------
__global__ void layernorm_bf16(const float* __restrict__ x,
                               const float* __restrict__ g,
                               const float* __restrict__ beta,
                               __bf16* __restrict__ out, int D) {
    const int row = blockIdx.x;
    const int tid = threadIdx.x;
    const float* xr = x + (size_t)row * D;
    float s = 0.0f, s2 = 0.0f;
    for (int i = tid; i < D; i += blockDim.x) {
        float v = xr[i];
        s += v; s2 += v * v;
    }
    __shared__ float r1[256], r2[256];
    r1[tid] = s; r2[tid] = s2;
    __syncthreads();
    for (int off = 128; off > 0; off >>= 1) {
        if (tid < off) { r1[tid] += r1[tid + off]; r2[tid] += r2[tid + off]; }
        __syncthreads();
    }
    const float mean = r1[0] / D;
    const float var  = r2[0] / D - mean * mean;
    const float rstd = rsqrtf(var + kEPS);
    for (int i = tid; i < D; i += blockDim.x) {
        float v = (xr[i] - mean) * rstd * g[i] + beta[i];
        out[(size_t)row * D + i] = (__bf16)v;
    }
}

// ---------------------------------------------------------------------------
// Weight packing.
// ---------------------------------------------------------------------------
// Wq/Wk/Wv (H,D,E) f32 -> Wqkvt bf16 [kNQ x D], row j: j<72 q, <144 k, <216 v.
__global__ void pack_qkvt(const float* __restrict__ Wq, const float* __restrict__ Wk,
                          const float* __restrict__ Wv, __bf16* __restrict__ out) {
    const int idx = blockIdx.x * blockDim.x + threadIdx.x;
    if (idx >= kNQ * kD) return;
    const int j = idx / kD, d = idx % kD;
    float v = 0.0f;
    if (j < 72)       { int h = j / kE,         e = j % kE;         v = Wq[((size_t)h * kD + d) * kE + e]; }
    else if (j < 144) { int h = (j - 72) / kE,  e = (j - 72) % kE;  v = Wk[((size_t)h * kD + d) * kE + e]; }
    else if (j < 216) { int h = (j - 144) / kE, e = (j - 144) % kE; v = Wv[((size_t)h * kD + d) * kE + e]; }
    out[idx] = (__bf16)v;
}

// in f32 [R x C] -> out bf16 [C x Kpad], zero-padding k in [R, Kpad).
__global__ void transpose_pack_bf16(const float* __restrict__ in, __bf16* __restrict__ out,
                                    int R, int C, int Kpad) {
    const int idx = blockIdx.x * blockDim.x + threadIdx.x;
    if (idx >= C * Kpad) return;
    const int n = idx / Kpad, k = idx % Kpad;
    out[idx] = (k < R) ? (__bf16)in[(size_t)k * C + n] : (__bf16)0.0f;
}

__global__ void zero_u32(unsigned int* __restrict__ p, int n) {
    const int idx = blockIdx.x * blockDim.x + threadIdx.x;
    if (idx < n) p[idx] = 0u;
}

// ---------------------------------------------------------------------------
// Causal attention, one (b,h,chunk-of-256-queries) per block; K/V in LDS.
// qkv: f32 [B*T, kNQ] (cols 0..71 q, 72..143 k, 144..215 v); obuf bf16 [B*T, kKO].
// ---------------------------------------------------------------------------
__global__ void attention_kernel(const float* __restrict__ qkv, __bf16* __restrict__ obuf) {
    constexpr int CHUNK = 256;
    const int chunks = kT / CHUNK;                 // 2
    const int bh = blockIdx.x / chunks;
    const int tc = blockIdx.x % chunks;
    const int b = bh / kH, h = bh % kH;
    const float scale = rsqrtf((float)kE);

    __shared__ float Ks[kT * kE];                  // 12 KB
    __shared__ float Vs[kT * kE];                  // 12 KB
    const float* base = qkv + (size_t)b * kT * kNQ;
    for (int i = threadIdx.x; i < kT * kE; i += blockDim.x) {
        const int s = i / kE, e = i % kE;
        Ks[i] = base[(size_t)s * kNQ + 72  + h * kE + e];
        Vs[i] = base[(size_t)s * kNQ + 144 + h * kE + e];
    }
    __syncthreads();

    const int t = tc * CHUNK + threadIdx.x;
    float q[kE];
    #pragma unroll
    for (int e = 0; e < kE; ++e)
        q[e] = base[(size_t)t * kNQ + h * kE + e] * scale;

    float m = -1e30f, l = 0.0f;
    float o[kE];
    #pragma unroll
    for (int e = 0; e < kE; ++e) o[e] = 0.0f;

    for (int s = 0; s <= t; ++s) {
        float sc = 0.0f;
        #pragma unroll
        for (int e = 0; e < kE; ++e) sc += q[e] * Ks[s * kE + e];
        const float mn = fmaxf(m, sc);
        const float alpha = __expf(m - mn);
        const float p = __expf(sc - mn);
        l = l * alpha + p;
        #pragma unroll
        for (int e = 0; e < kE; ++e) o[e] = o[e] * alpha + p * Vs[s * kE + e];
        m = mn;
    }
    const float inv = 1.0f / l;
    __bf16* orow = obuf + (size_t)(b * kT + t) * kKO + h * kE;
    #pragma unroll
    for (int e = 0; e < kE; ++e) orow[e] = (__bf16)(o[e] * inv);
}

// ---------------------------------------------------------------------------
// Launch
// ---------------------------------------------------------------------------
extern "C" void kernel_launch(void* const* d_in, const int* in_sizes, int n_in,
                              void* d_out, int out_size, void* d_ws, size_t ws_size,
                              hipStream_t stream) {
    const float* x     = (const float*)d_in[0];
    const float* Wq    = (const float*)d_in[1];
    const float* Wk    = (const float*)d_in[2];
    const float* Wv    = (const float*)d_in[3];
    const float* Wo    = (const float*)d_in[4];
    const float* bo    = (const float*)d_in[5];
    const float* W1    = (const float*)d_in[6];
    const float* b1    = (const float*)d_in[7];
    const float* W2    = (const float*)d_in[8];
    const float* b2    = (const float*)d_in[9];
    const float* g1    = (const float*)d_in[10];
    const float* beta1 = (const float*)d_in[11];
    const float* g2    = (const float*)d_in[12];
    const float* beta2 = (const float*)d_in[13];
    (void)in_sizes; (void)n_in; (void)out_size; (void)ws_size;

    char* ws = (char*)d_ws;
    size_t off = 0;
    auto alloc = [&](size_t bytes) -> void* {
        void* p = ws + off;
        off = (off + bytes + 255) & ~(size_t)255;
        return p;
    };

    __bf16* xn    = (__bf16*)alloc((size_t)kBT * kD * 2);     // LN1(x) bf16
    __bf16* wqkvt = (__bf16*)alloc((size_t)kNQ * kD * 2);     // [256 x 768]
    __bf16* wot   = (__bf16*)alloc((size_t)kD * kKO * 2);     // [768 x 96]
    __bf16* w1t   = (__bf16*)alloc((size_t)kDFF * kD * 2);    // [3072 x 768]
    __bf16* w2t   = (__bf16*)alloc((size_t)kD * kDFF * 2);    // [768 x 3072]
    float*  qkv   = (float*) alloc((size_t)kBT * kNQ * 4);    // q|k|v f32
    __bf16* obuf  = (__bf16*)alloc((size_t)kBT * kKO * 2);    // attn out bf16 (padded)
    float*  x1    = (float*) alloc((size_t)kBT * kD * 4);     // x + attn proj
    __bf16* ybuf  = (__bf16*)alloc((size_t)kBT * kD * 2);     // LN2(x1) bf16
    __bf16* hbuf  = (__bf16*)alloc((size_t)kBT * kDFF * 2);   // relu(ffn1) bf16

    // --- weight packing (bf16, transposed) ---
    {
        int n = kNQ * kD;
        pack_qkvt<<<(n + 255) / 256, 256, 0, stream>>>(Wq, Wk, Wv, wqkvt);
    }
    {
        int n = kD * kKO;
        transpose_pack_bf16<<<(n + 255) / 256, 256, 0, stream>>>(Wo, wot, 72, kD, kKO);
    }
    {
        int n = kDFF * kD;
        transpose_pack_bf16<<<(n + 255) / 256, 256, 0, stream>>>(W1, w1t, kD, kDFF, kD);
    }
    {
        int n = kD * kDFF;
        transpose_pack_bf16<<<(n + 255) / 256, 256, 0, stream>>>(W2, w2t, kDFF, kD, kDFF);
    }

    // --- LN1 ---
    layernorm_bf16<<<kBT, 256, 0, stream>>>(x, g1, beta1, xn, kD);

    // --- QKV projection: [BT x 768] x [768 x 256] -> qkv f32 ---
    gemm_bf16_wmma<false, false, false, false>
        <<<dim3(kBT / 128, kNQ / 64), 128, 0, stream>>>(
        xn, wqkvt, nullptr, nullptr, qkv, nullptr, kBT, kNQ, kD);

    // --- attention ---
    {
        int n = (kBT * kKO) / 2;  // bf16 pairs as u32
        zero_u32<<<(n + 255) / 256, 256, 0, stream>>>((unsigned int*)obuf, n);
    }
    attention_kernel<<<kB * kH * (kT / 256), 256, 0, stream>>>(qkv, obuf);

    // --- out projection + residual: x1 = x + o @ Wo + bo ---
    gemm_bf16_wmma<true, true, false, false>
        <<<dim3(kBT / 128, kD / 64), 128, 0, stream>>>(
        obuf, wot, bo, x, x1, nullptr, kBT, kD, kKO);

    // --- LN2 ---
    layernorm_bf16<<<kBT, 256, 0, stream>>>(x1, g2, beta2, ybuf, kD);

    // --- FFN1: relu(y @ W1 + b1) -> bf16 ---
    gemm_bf16_wmma<true, false, true, true>
        <<<dim3(kBT / 128, kDFF / 64), 128, 0, stream>>>(
        ybuf, w1t, b1, nullptr, nullptr, hbuf, kBT, kDFF, kD);

    // --- FFN2 + residual: out = x1 + h @ W2 + b2 ---
    gemm_bf16_wmma<true, true, false, false>
        <<<dim3(kBT / 128, kD / 64), 128, 0, stream>>>(
        hbuf, w2t, b2, x1, (float*)d_out, nullptr, kBT, kD, kDFF);
}
